// TokenMixerLargeBlock_13262859010555
// MI455X (gfx1250) — compile-verified, matrix operand order checked by reading, not run on hardware
//
#include <hip/hip_runtime.h>
#include <hip/hip_bf16.h>

typedef __attribute__((ext_vector_type(16))) __bf16 v16bf;
typedef __attribute__((ext_vector_type(8)))  __bf16 v8bf;
typedef __attribute__((ext_vector_type(8)))  float  v8f;

#define AS_STR 136   // 128 K elems + 8 pad (bf16) -> 272B row stride, 16B aligned
#define YS_STR 132   // f32 row stride for epilogue overlay

__device__ __forceinline__ __bf16 tobf(float f) { return (__bf16)f; }

// Load one 16-lane-row fragment per the CDNA5 16-bit A/B 16x32 VGPR layout:
// half 0 -> K kk+0..7 then kk+16..23 ; half 1 -> K kk+8..15 then kk+24..31
__device__ __forceinline__ v16bf load_frag(const __bf16* __restrict__ row, int kk, int hf) {
  v8bf lo = *(const v8bf*)(row + kk + 8 * hf);
  v8bf hi = *(const v8bf*)(row + kk + 16 + 8 * hf);
  v16bf r;
#pragma unroll
  for (int i = 0; i < 8; ++i) { r[i] = lo[i]; r[i + 8] = hi[i]; }
  return r;
}

__device__ __forceinline__ float silu_f(float g) { return g / (1.0f + __expf(-g)); }

// -----------------------------------------------------------------------------
// Kernel 1: per-token up & gate projections + SiLU-mul, h[t][b][e] to workspace.
// A[b][k] = Asrc[(b*128+k)*128 + tok]  (token-mix transpose folded into load)
// grid = (E/64 = 8, T = 128), block = 256 (8 waves)
// -----------------------------------------------------------------------------
__global__ __launch_bounds__(256)
void upgate_kernel(const float* __restrict__ Asrc, const float* __restrict__ Wu,
                   const float* __restrict__ bu,   const float* __restrict__ Wg,
                   const float* __restrict__ bg,   float* __restrict__ hws) {
  __shared__ __bf16 As[64 * AS_STR];
  __shared__ __bf16 Bu[64 * AS_STR];
  __shared__ __bf16 Bg[64 * AS_STR];

  const int tid = threadIdx.x;
  const int tok = blockIdx.y;
  const int n0  = blockIdx.x * 64;

  // Stage A: 64 rows (batch) x 128 K (feature), bf16
  for (int idx = tid; idx < 64 * 128; idx += 256) {
    int b = idx >> 7, k = idx & 127;
    As[b * AS_STR + k] = tobf(Asrc[(b * 128 + k) * 128 + tok]);
  }
  // Stage Wu/Wg tile transposed: B[e][k] = W[tok][k][n0+e]
  for (int idx = tid; idx < 64 * 128; idx += 256) {
    int k = idx >> 6, e = idx & 63;
    size_t g = (size_t)(tok * 128 + k) * 512 + n0 + e;
    Bu[e * AS_STR + k] = tobf(Wu[g]);
    Bg[e * AS_STR + k] = tobf(Wg[g]);
  }
  __syncthreads();

  const int w = tid >> 5, lane = tid & 31, ln = lane & 15, hf = lane >> 4;
  const int mi = w >> 1;                     // both of this wave's tiles share mi
  v8f cu[2] = {}, cg[2] = {};
  const __bf16* arow = As + (mi * 16 + ln) * AS_STR;

#pragma unroll
  for (int kk = 0; kk < 128; kk += 32) {
    v16bf a = load_frag(arow, kk, hf);       // reused across 2 N-tiles x 2 mats
#pragma unroll
    for (int j = 0; j < 2; ++j) {
      int ni = (w & 1) * 2 + j;
      v16bf bU = load_frag(Bu + (ni * 16 + ln) * AS_STR, kk, hf);
      v16bf bG = load_frag(Bg + (ni * 16 + ln) * AS_STR, kk, hf);
      cu[j] = __builtin_amdgcn_wmma_f32_16x16x32_bf16(false, a, false, bU,
                                                      (short)0, cu[j], false, false);
      cg[j] = __builtin_amdgcn_wmma_f32_16x16x32_bf16(false, a, false, bG,
                                                      (short)0, cg[j], false, false);
    }
  }

  // Register-local epilogue: h = (up+bu) * silu(gate+bg)
#pragma unroll
  for (int j = 0; j < 2; ++j) {
    int ni = (w & 1) * 2 + j;
    int e  = n0 + ni * 16 + ln;
    float bu_v = bu[tok * 512 + e];
    float bg_v = bg[tok * 512 + e];
#pragma unroll
    for (int r = 0; r < 8; ++r) {
      int b = mi * 16 + r + 8 * hf;          // C/D layout: M = r + 8*half, N = lane%16
      float u = cu[j][r] + bu_v;
      float g = cg[j][r] + bg_v;
      hws[(size_t)(tok * 64 + b) * 512 + e] = u * silu_f(g);
    }
  }
}

// -----------------------------------------------------------------------------
// Kernel 2: per-token down projection + bias + residual + RMS-norm.
// y[64x128] = h_t[64x512] @ Wd[t][512x128]; out = rmsnorm(y + A) * scale
// grid = (T = 128), block = 256 (8 waves); wave w owns N-block w (B reuse x4)
// -----------------------------------------------------------------------------
__global__ __launch_bounds__(256)
void down_kernel(const float* __restrict__ h,    const float* __restrict__ Wd,
                 const float* __restrict__ bd,   const float* __restrict__ Asrc,
                 const float* __restrict__ scale, float* __restrict__ outp) {
  __shared__ __align__(16) unsigned char smem[(64 + 128) * AS_STR * 2]; // 52224 B
  __bf16* As = (__bf16*)smem;               // 64  x AS_STR (GEMM phase)
  __bf16* Bs = As + 64 * AS_STR;            // 128 x AS_STR (GEMM phase)
  float*  ys = (float*)smem;                // 64 x YS_STR  (epilogue overlay)
  float*  red = (float*)(smem + 64 * YS_STR * 4); // 256 partials + 64 rrms

  const int tid = threadIdx.x;
  const int tok = blockIdx.x;
  const int w = tid >> 5, lane = tid & 31, ln = lane & 15, hf = lane >> 4;

  v8f c[4] = {};
  for (int kc = 0; kc < 4; ++kc) {
    const int K0 = kc * 128;
    __syncthreads();                         // previous chunk fully consumed
    for (int idx = tid; idx < 64 * 128; idx += 256) {
      int b = idx >> 7, k = idx & 127;
      As[b * AS_STR + k] = tobf(h[(size_t)(tok * 64 + b) * 512 + K0 + k]);
    }
    for (int idx = tid; idx < 128 * 128; idx += 256) {
      int k = idx >> 7, n = idx & 127;
      Bs[n * AS_STR + k] = tobf(Wd[(size_t)(tok * 512 + K0 + k) * 128 + n]);
    }
    if (kc < 3)                              // warm L2 for next weight chunk
      __builtin_prefetch(&Wd[(size_t)(tok * 512 + K0 + 128) * 128 + tid], 0, 1);
    __syncthreads();

    const __bf16* brow = Bs + (w * 16 + ln) * AS_STR;
#pragma unroll
    for (int kk = 0; kk < 128; kk += 32) {
      v16bf bfrag = load_frag(brow, kk, hf); // reused across 4 M-tiles
#pragma unroll
      for (int mi = 0; mi < 4; ++mi) {
        v16bf a = load_frag(As + (mi * 16 + ln) * AS_STR, kk, hf);
        c[mi] = __builtin_amdgcn_wmma_f32_16x16x32_bf16(false, a, false, bfrag,
                                                        (short)0, c[mi], false, false);
      }
    }
  }
  __syncthreads();

  // bias + residual (residual == A-gather of this layer) into LDS overlay
  {
    int j = w * 16 + ln;
    float bd_v = bd[tok * 128 + j];
#pragma unroll
    for (int mi = 0; mi < 4; ++mi) {
#pragma unroll
      for (int r = 0; r < 8; ++r) {
        int b = mi * 16 + r + 8 * hf;
        float v = c[mi][r] + bd_v + Asrc[(b * 128 + j) * 128 + tok];
        ys[b * YS_STR + j] = v;
      }
    }
  }
  __syncthreads();

  // RMS-norm: 4-way partial sum of squares per row, then combine
  {
    int row = tid >> 2, part = tid & 3;
    const float* yr = ys + row * YS_STR + part * 32;
    float s = 0.f;
#pragma unroll
    for (int i = 0; i < 32; ++i) s += yr[i] * yr[i];
    red[row * 4 + part] = s;
  }
  __syncthreads();
  if (tid < 64) {
    float s = red[tid * 4] + red[tid * 4 + 1] + red[tid * 4 + 2] + red[tid * 4 + 3];
    red[256 + tid] = rsqrtf(s * (1.0f / 128.0f) + 1e-5f);
  }
  __syncthreads();
  for (int idx = tid; idx < 64 * 128; idx += 256) {
    int b = idx >> 7, j = idx & 127;
    outp[(b * 128 + tok) * 128 + j] = ys[b * YS_STR + j] * red[256 + b] * scale[j];
  }
}

// -----------------------------------------------------------------------------
extern "C" void kernel_launch(void* const* d_in, const int* in_sizes, int n_in,
                              void* d_out, int out_size, void* d_ws, size_t ws_size,
                              hipStream_t stream) {
  const float* x   = (const float*)d_in[0];
  const float* W1u = (const float*)d_in[1];
  const float* b1u = (const float*)d_in[2];
  const float* W1g = (const float*)d_in[3];
  const float* b1g = (const float*)d_in[4];
  const float* W1d = (const float*)d_in[5];
  const float* b1d = (const float*)d_in[6];
  const float* s1  = (const float*)d_in[7];
  const float* W2u = (const float*)d_in[8];
  const float* b2u = (const float*)d_in[9];
  const float* W2g = (const float*)d_in[10];
  const float* b2g = (const float*)d_in[11];
  const float* W2d = (const float*)d_in[12];
  const float* b2d = (const float*)d_in[13];
  const float* s2  = (const float*)d_in[14];

  float* hws = (float*)d_ws;                          // [128][64][512] f32, 16.8 MB
  float* x1s = hws + (size_t)128 * 64 * 512;          // [64][128][128] f32,  4.2 MB
  float* out = (float*)d_out;

  dim3 gUG(8, 128);
  // Layer 1 (A-source = x, token-mix transpose folded into gathers)
  upgate_kernel<<<gUG, 256, 0, stream>>>(x,   W1u, b1u, W1g, b1g, hws);
  down_kernel<<<128, 256, 0, stream>>>(hws, W1d, b1d, x,   s1, x1s);
  // Layer 2 (A-source = x1; reverse token-mix is the same transpose gather)
  upgate_kernel<<<gUG, 256, 0, stream>>>(x1s, W2u, b2u, W2g, b2g, hws);
  down_kernel<<<128, 256, 0, stream>>>(hws, W2d, b2d, x1s, s2, out);
}
